// EdgeNetVAE_2731599200743
// MI455X (gfx1250) — matrix operand order repaired and checked by reading, not compiled, not run on hardware
//
#include <hip/hip_runtime.h>
#include <hip/hip_bf16.h>

// ---------------------------------------------------------------------------
// EdgeNet VAE on MI455X (gfx1250).
// Per-edge 32x32 MLP GEMMs run on v_wmma_f32_16x16x32_f16, 16 edges per wave,
// persistent grid-stride so weight B-tiles stay resident in VGPRs.
// Layer-1 of each EdgeConv is factored into per-node partials P/Q (linear
// before ReLU): per-edge work = gather + v_pk_add_f16 + WMMA + f32 atomics.
// Decoder re-layouts D(f32)->A(f16) through wave-private LDS, ordered with
// inline "s_wait_dscnt 0x0" (same-wave LDS is in-order; no block barrier).
// ---------------------------------------------------------------------------

typedef __attribute__((ext_vector_type(16))) _Float16 v16h;
typedef __attribute__((ext_vector_type(8)))  _Float16 h8;
typedef __attribute__((ext_vector_type(8)))  float    v8f;

__device__ __forceinline__ h8 relu_h8(h8 v) {
#pragma unroll
  for (int i = 0; i < 8; ++i) v[i] = (v[i] > (_Float16)0) ? v[i] : (_Float16)0;
  return v;
}

__device__ __forceinline__ void wait_lds() {
  // Same-wave LDS RAW: DS ops are in-order per wave; the explicit wait +
  // memory clobber also stops the compiler from reordering around it.
  asm volatile("s_wait_dscnt 0x0" ::: "memory");
}

// ---------------------------- utility kernels ------------------------------

__global__ void fill0_kernel(float* __restrict__ p, int n) {
  int i = blockIdx.x * blockDim.x + threadIdx.x;
  if (i < n) p[i] = 0.0f;
}

// Convert 32x32 (and padded 32x4) f32 weights into WMMA B-operand f16 tiles.
// Per 32x16 tile: slot = lane*16 + j holds W[k][n] with k=(j>>3)*16+(lane>>4)*8+(j&7),
// n=lane&15 -> a v16h load at (tile + lane*16) is the B operand layout.
__global__ void prep_weights_kernel(const float* __restrict__ encW2,
                                    const float* __restrict__ decW2,
                                    const float* __restrict__ decW3,
                                    _Float16* __restrict__ encBt,
                                    _Float16* __restrict__ decBt,
                                    _Float16* __restrict__ dec3Bt) {
  int t = threadIdx.x;                 // 0..511
  if (t >= 512) return;
  int lane = t >> 4;
  int j    = t & 15;
  int k    = ((j >> 3) << 4) + ((lane >> 4) << 3) + (j & 7);   // K index 0..31
  int n    = lane & 15;                // N index 0..15
  encBt[t]       = (_Float16)encW2[k * 32 + n];
  encBt[512 + t] = (_Float16)encW2[k * 32 + 16 + n];
  decBt[t]       = (_Float16)decW2[k * 32 + n];
  decBt[512 + t] = (_Float16)decW2[k * 32 + 16 + n];
  dec3Bt[t]      = (n < 4) ? (_Float16)decW3[k * 4 + n] : (_Float16)0;
}

// Per-node encoder layer-1 partials:
//   P[i] = x_i @ (W1[0:4] - W1[4:8]) + b1   (role: destination)
//   Q[i] = x_i @  W1[4:8]                   (role: source)
__global__ void enc_node_kernel(const float* __restrict__ x,
                                const float* __restrict__ W1,
                                const float* __restrict__ b1,
                                _Float16* __restrict__ P,
                                _Float16* __restrict__ Q, int N) {
  int i = blockIdx.x * blockDim.x + threadIdx.x;
  if (i >= N) return;
  float xv[4];
#pragma unroll
  for (int k = 0; k < 4; ++k) xv[k] = x[(size_t)i * 4 + k];
#pragma unroll
  for (int o = 0; o < 32; ++o) {
    float top = 0.f, bot = 0.f;
#pragma unroll
    for (int k = 0; k < 4; ++k) {
      top += xv[k] * W1[k * 32 + o];
      bot += xv[k] * W1[(k + 4) * 32 + o];
    }
    P[(size_t)i * 32 + o] = (_Float16)(top - bot + b1[o]);
    Q[(size_t)i * 32 + o] = (_Float16)bot;
  }
}

__global__ void count_kernel(const int* __restrict__ dst, float* __restrict__ cnt, int E) {
  int e = blockIdx.x * blockDim.x + threadIdx.x;
  if (e < E) atomicAdd(&cnt[dst[e]], 1.0f);
}

// ------------------------- encoder edge kernel -----------------------------
// Persistent waves; each loop iteration handles one 16-edge tile:
// A = relu(P[dst] + Q[src]) (16x32 f16), two WMMAs vs resident encBt tiles,
// relu(+b2), branch-free atomic segment-sum (row dst via global_load_b128).
__global__ void __launch_bounds__(256)
enc_edge_kernel(const int* __restrict__ srcArr, const int* __restrict__ dstArr,
                const _Float16* __restrict__ P, const _Float16* __restrict__ Q,
                const _Float16* __restrict__ encBt, const float* __restrict__ b2,
                float* __restrict__ sumEnc, int tilesFull) {
  const int lane  = threadIdx.x & 31;
  const int m     = lane & 15;
  const int hi    = lane >> 4;
  const int wave0 = (blockIdx.x * blockDim.x + threadIdx.x) >> 5;
  const int nWav  = (gridDim.x * blockDim.x) >> 5;

  // Weight tiles + biases stay in VGPRs for all tiles this wave processes.
  const v16h B0 = *(const v16h*)(encBt + lane * 16);
  const v16h B1 = *(const v16h*)(encBt + 512 + lane * 16);
  const float bias0 = b2[m], bias1 = b2[16 + m];
  const v8f cz = {};

  for (int tile = wave0; tile < tilesFull; tile += nWav) {
    const int e0 = tile << 4;
    if (tile + nWav < tilesFull) {            // prefetch next tile's indices
      const int en = e0 + (nWav << 4);
      __builtin_prefetch(dstArr + en, 0, 1);
      __builtin_prefetch(srcArr + en, 0, 1);
    }

    const int e = e0 + m;
    const int d = dstArr[e];
    const int s = srcArr[e];

    // A layout: lane m+16*hi holds K chunks [8*hi,8*hi+8) and [16+8*hi, ...)
    h8 a0 = *(const h8*)(P + (size_t)d * 32 + hi * 8)
          + *(const h8*)(Q + (size_t)s * 32 + hi * 8);
    h8 a1 = *(const h8*)(P + (size_t)d * 32 + 16 + hi * 8)
          + *(const h8*)(Q + (size_t)s * 32 + 16 + hi * 8);
    a0 = relu_h8(a0);
    a1 = relu_h8(a1);
    v16h A;
#pragma unroll
    for (int i = 0; i < 8; ++i) { A[i] = a0[i]; A[8 + i] = a1[i]; }

    v8f d0 = __builtin_amdgcn_wmma_f32_16x16x32_f16(false, A, false, B0, (short)0, cz, false, false);
    v8f d1 = __builtin_amdgcn_wmma_f32_16x16x32_f16(false, A, false, B1, (short)0, cz, false, false);

    // Row destinations for this lane's 8 D-rows (rows 8*hi + r): two b128 loads.
    const int rbase = e0 + (hi << 3);
    const int4 dv0 = *(const int4*)(dstArr + rbase);
    const int4 dv1 = *(const int4*)(dstArr + rbase + 4);
    const int dr[8] = {dv0.x, dv0.y, dv0.z, dv0.w, dv1.x, dv1.y, dv1.z, dv1.w};
#pragma unroll
    for (int r = 0; r < 8; ++r) {
      const float v0 = fmaxf(d0[r] + bias0, 0.f);
      const float v1 = fmaxf(d1[r] + bias1, 0.f);
      const size_t base = (size_t)dr[r] * 32;
      atomicAdd(&sumEnc[base + m],      v0);
      atomicAdd(&sumEnc[base + 16 + m], v1);
    }
  }
}

// Per-node: h = sum/max(cnt,1); mu/logvar heads; z = mu + eps*exp(0.5*lv);
// then decoder layer-1 partials Pd/Qd from z (same factoring as encoder).
__global__ void node_update_kernel(const float* __restrict__ sumEnc,
                                   const float* __restrict__ cnt,
                                   const float* __restrict__ eps,
                                   const float* __restrict__ muW, const float* __restrict__ muB,
                                   const float* __restrict__ varW, const float* __restrict__ varB,
                                   const float* __restrict__ dW1, const float* __restrict__ db1,
                                   float* __restrict__ outMu, float* __restrict__ outLV,
                                   _Float16* __restrict__ Pd, _Float16* __restrict__ Qd, int N) {
  int i = blockIdx.x * blockDim.x + threadIdx.x;
  if (i >= N) return;
  float inv = 1.0f / fmaxf(cnt[i], 1.0f);
  float mu0 = muB[0], mu1 = muB[1], lv0 = varB[0], lv1 = varB[1];
#pragma unroll 8
  for (int k = 0; k < 32; ++k) {
    float h = sumEnc[(size_t)i * 32 + k] * inv;
    mu0 += h * muW[k * 2 + 0]; mu1 += h * muW[k * 2 + 1];
    lv0 += h * varW[k * 2 + 0]; lv1 += h * varW[k * 2 + 1];
  }
  float z0 = mu0 + eps[(size_t)i * 2 + 0] * __expf(0.5f * lv0);
  float z1 = mu1 + eps[(size_t)i * 2 + 1] * __expf(0.5f * lv1);
  outMu[(size_t)i * 2 + 0] = mu0; outMu[(size_t)i * 2 + 1] = mu1;
  outLV[(size_t)i * 2 + 0] = lv0; outLV[(size_t)i * 2 + 1] = lv1;
#pragma unroll
  for (int o = 0; o < 32; ++o) {
    float top = z0 * dW1[0 * 32 + o] + z1 * dW1[1 * 32 + o];
    float bot = z0 * dW1[2 * 32 + o] + z1 * dW1[3 * 32 + o];
    Pd[(size_t)i * 32 + o] = (_Float16)(top - bot + db1[o]);
    Qd[(size_t)i * 32 + o] = (_Float16)bot;
  }
}

// ------------------------- decoder edge kernel -----------------------------
// Hidden 32x32 on WMMA -> relu -> f16 tile in wave-private LDS (re-layout
// D->A) -> WMMA vs padded W3 -> stage 16x4 outputs + row dsts in LDS ->
// all 32 lanes retire exactly 2 atomics each (balanced).
__global__ void __launch_bounds__(256)
dec_edge_kernel(const int* __restrict__ srcArr, const int* __restrict__ dstArr,
                const _Float16* __restrict__ Pd, const _Float16* __restrict__ Qd,
                const _Float16* __restrict__ decBt, const float* __restrict__ db2,
                const _Float16* __restrict__ dec3Bt, const float* __restrict__ db3,
                float* __restrict__ sumDec, int tilesFull) {
  __shared__ _Float16 hTile[8][16 * 32];  // per-wave 16x32 f16 hidden tile
  __shared__ float    oTile[8][64];       // per-wave 16 edges x 4 channels
  __shared__ int      dTile[8][16];       // per-wave row destinations

  const int wib   = threadIdx.x >> 5;
  const int lane  = threadIdx.x & 31;
  const int m     = lane & 15;
  const int hi    = lane >> 4;
  const int wave0 = (blockIdx.x * blockDim.x + threadIdx.x) >> 5;
  const int nWav  = (gridDim.x * blockDim.x) >> 5;

  const v16h B0 = *(const v16h*)(decBt + lane * 16);
  const v16h B1 = *(const v16h*)(decBt + 512 + lane * 16);
  const v16h B3 = *(const v16h*)(dec3Bt + lane * 16);
  const float bias0 = db2[m], bias1 = db2[16 + m];
  const float b3c   = db3[lane & 3];
  const v8f cz = {};

  _Float16* t  = hTile[wib];
  float*    fo = oTile[wib];
  int*      fd = dTile[wib];

  for (int tile = wave0; tile < tilesFull; tile += nWav) {
    const int e0 = tile << 4;
    if (tile + nWav < tilesFull) {
      const int en = e0 + (nWav << 4);
      __builtin_prefetch(dstArr + en, 0, 1);
      __builtin_prefetch(srcArr + en, 0, 1);
    }

    const int e = e0 + m;
    const int d = dstArr[e];
    const int s = srcArr[e];

    h8 a0 = *(const h8*)(Pd + (size_t)d * 32 + hi * 8)
          + *(const h8*)(Qd + (size_t)s * 32 + hi * 8);
    h8 a1 = *(const h8*)(Pd + (size_t)d * 32 + 16 + hi * 8)
          + *(const h8*)(Qd + (size_t)s * 32 + 16 + hi * 8);
    a0 = relu_h8(a0);
    a1 = relu_h8(a1);
    v16h A;
#pragma unroll
    for (int i = 0; i < 8; ++i) { A[i] = a0[i]; A[8 + i] = a1[i]; }

    v8f d0 = __builtin_amdgcn_wmma_f32_16x16x32_f16(false, A, false, B0, (short)0, cz, false, false);
    v8f d1 = __builtin_amdgcn_wmma_f32_16x16x32_f16(false, A, false, B1, (short)0, cz, false, false);

    // Stage relu(h@W2+b2) as f16 row-major [edge_row][channel].
#pragma unroll
    for (int r = 0; r < 8; ++r) {
      const float v0 = fmaxf(d0[r] + bias0, 0.f);
      const float v1 = fmaxf(d1[r] + bias1, 0.f);
      t[(8 * hi + r) * 32 + m]      = (_Float16)v0;
      t[(8 * hi + r) * 32 + 16 + m] = (_Float16)v1;
    }
    wait_lds();

    // Re-read in A-operand layout (ds_load_b128), project through padded W3.
    h8 c0 = *(const h8*)(t + m * 32 + hi * 8);
    h8 c1 = *(const h8*)(t + m * 32 + 16 + hi * 8);
    v16h A2;
#pragma unroll
    for (int i = 0; i < 8; ++i) { A2[i] = c0[i]; A2[8 + i] = c1[i]; }
    v8f d2 = __builtin_amdgcn_wmma_f32_16x16x32_f16(false, A2, false, B3, (short)0, cz, false, false);

    // Stage the 16x4 real outputs and the 16 row destinations.
    if (m < 4) {
#pragma unroll
      for (int r = 0; r < 8; ++r) fo[(8 * hi + r) * 4 + m] = d2[r];
    }
    if (m == 4) {
      const int rbase = e0 + (hi << 3);
      const int4 dv0 = *(const int4*)(dstArr + rbase);
      const int4 dv1 = *(const int4*)(dstArr + rbase + 4);
      fd[hi * 8 + 0] = dv0.x; fd[hi * 8 + 1] = dv0.y;
      fd[hi * 8 + 2] = dv0.z; fd[hi * 8 + 3] = dv0.w;
      fd[hi * 8 + 4] = dv1.x; fd[hi * 8 + 5] = dv1.y;
      fd[hi * 8 + 6] = dv1.z; fd[hi * 8 + 7] = dv1.w;
    }
    wait_lds();

    // Balanced retire: every lane handles (row, ch) and (row+8, ch).
    const int ch  = lane & 3;
    const int row = lane >> 2;          // 0..7
    const float v0 = fo[row * 4 + ch] + b3c;
    const float v1 = fo[(row + 8) * 4 + ch] + b3c;
    const int dr0 = fd[row];
    const int dr1 = fd[row + 8];
    atomicAdd(&sumDec[(size_t)dr0 * 4 + ch], v0);
    atomicAdd(&sumDec[(size_t)dr1 * 4 + ch], v1);
  }
}

// ------------------------ scalar tail kernels ------------------------------
// Handle E % 16 leftover edges (never launched for E = 3.2M).

__global__ void enc_tail_kernel(const int* __restrict__ srcArr, const int* __restrict__ dstArr,
                                const _Float16* __restrict__ P, const _Float16* __restrict__ Q,
                                const float* __restrict__ W2, const float* __restrict__ b2,
                                float* __restrict__ sumEnc, int eStart, int E) {
  int e = eStart + blockIdx.x * blockDim.x + threadIdx.x;
  if (e >= E) return;
  int d = dstArr[e], s = srcArr[e];
  float act[32];
#pragma unroll
  for (int k = 0; k < 32; ++k)
    act[k] = fmaxf((float)P[(size_t)d * 32 + k] + (float)Q[(size_t)s * 32 + k], 0.f);
  for (int n = 0; n < 32; ++n) {
    float acc = b2[n];
#pragma unroll
    for (int k = 0; k < 32; ++k) acc += act[k] * W2[k * 32 + n];
    atomicAdd(&sumEnc[(size_t)d * 32 + n], fmaxf(acc, 0.f));
  }
}

__global__ void dec_tail_kernel(const int* __restrict__ srcArr, const int* __restrict__ dstArr,
                                const _Float16* __restrict__ Pd, const _Float16* __restrict__ Qd,
                                const float* __restrict__ W2, const float* __restrict__ b2,
                                const float* __restrict__ W3, const float* __restrict__ b3,
                                float* __restrict__ sumDec, int eStart, int E) {
  int e = eStart + blockIdx.x * blockDim.x + threadIdx.x;
  if (e >= E) return;
  int d = dstArr[e], s = srcArr[e];
  float act[32], h2[32];
#pragma unroll
  for (int k = 0; k < 32; ++k)
    act[k] = fmaxf((float)Pd[(size_t)d * 32 + k] + (float)Qd[(size_t)s * 32 + k], 0.f);
  for (int n = 0; n < 32; ++n) {
    float acc = b2[n];
#pragma unroll
    for (int k = 0; k < 32; ++k) acc += act[k] * W2[k * 32 + n];
    h2[n] = fmaxf(acc, 0.f);
  }
  for (int c = 0; c < 4; ++c) {
    float acc = b3[c];
#pragma unroll
    for (int k = 0; k < 32; ++k) acc += h2[k] * W3[k * 4 + c];
    atomicAdd(&sumDec[(size_t)d * 4 + c], acc);
  }
}

__global__ void finalize_kernel(const float* __restrict__ sumDec,
                                const float* __restrict__ cnt,
                                float* __restrict__ out, int N) {
  int idx = blockIdx.x * blockDim.x + threadIdx.x;
  if (idx >= N * 4) return;
  out[idx] = sumDec[idx] / fmaxf(cnt[idx >> 2], 1.0f);
}

// ------------------------------- launcher ----------------------------------

extern "C" void kernel_launch(void* const* d_in, const int* in_sizes, int n_in,
                              void* d_out, int out_size, void* d_ws, size_t ws_size,
                              hipStream_t stream) {
  const float* x     = (const float*)d_in[0];
  const int*   ei    = (const int*)  d_in[1];
  const float* eps   = (const float*)d_in[2];
  const float* encW1 = (const float*)d_in[3];
  const float* encB1 = (const float*)d_in[4];
  const float* encW2 = (const float*)d_in[5];
  const float* encB2 = (const float*)d_in[6];
  const float* muW   = (const float*)d_in[7];
  const float* muB   = (const float*)d_in[8];
  const float* varW  = (const float*)d_in[9];
  const float* varB  = (const float*)d_in[10];
  const float* decW1 = (const float*)d_in[11];
  const float* decB1 = (const float*)d_in[12];
  const float* decW2 = (const float*)d_in[13];
  const float* decB2 = (const float*)d_in[14];
  const float* decW3 = (const float*)d_in[15];
  const float* decB3 = (const float*)d_in[16];
  (void)n_in; (void)out_size; (void)ws_size;

  const int N = in_sizes[0] / 4;
  const int E = in_sizes[1] / 2;
  const int* srcArr = ei;       // edge_index[0]
  const int* dstArr = ei + E;   // edge_index[1]

  // Workspace carving (256B aligned slices).
  char* ws = (char*)d_ws;
  size_t off = 0;
  auto take = [&](size_t bytes) -> char* {
    char* p = ws + off;
    off = (off + bytes + 255) & ~(size_t)255;
    return p;
  };
  float*    sumEnc = (float*)   take((size_t)N * 32 * sizeof(float));
  float*    cnt    = (float*)   take((size_t)N * sizeof(float));
  float*    sumDec = (float*)   take((size_t)N * 4 * sizeof(float));
  _Float16* P      = (_Float16*)take((size_t)N * 32 * sizeof(_Float16));
  _Float16* Q      = (_Float16*)take((size_t)N * 32 * sizeof(_Float16));
  _Float16* Pd     = (_Float16*)take((size_t)N * 32 * sizeof(_Float16));
  _Float16* Qd     = (_Float16*)take((size_t)N * 32 * sizeof(_Float16));
  _Float16* encBt  = (_Float16*)take(1024 * sizeof(_Float16));
  _Float16* decBt  = (_Float16*)take(1024 * sizeof(_Float16));
  _Float16* dec3Bt = (_Float16*)take(512 * sizeof(_Float16));

  float* outO  = (float*)d_out;              // [N,4]
  float* outMu = outO + (size_t)N * 4;       // [N,2]
  float* outLV = outMu + (size_t)N * 2;      // [N,2]

  const int tilesFull  = E / 16;             // full 16-edge WMMA tiles
  const int eTail      = tilesFull * 16;
  const int rem        = E - eTail;
  int edgeBlocks = (tilesFull + 7) / 8;      // 8 waves (tiles) per 256-thread block
  if (edgeBlocks > 1024) edgeBlocks = 1024;  // persistent grid-stride

  fill0_kernel<<<(N * 32 + 255) / 256, 256, 0, stream>>>(sumEnc, N * 32);
  fill0_kernel<<<(N + 255) / 256, 256, 0, stream>>>(cnt, N);
  fill0_kernel<<<(N * 4 + 255) / 256, 256, 0, stream>>>(sumDec, N * 4);
  prep_weights_kernel<<<1, 512, 0, stream>>>(encW2, decW2, decW3, encBt, decBt, dec3Bt);
  enc_node_kernel<<<(N + 255) / 256, 256, 0, stream>>>(x, encW1, encB1, P, Q, N);
  count_kernel<<<(E + 255) / 256, 256, 0, stream>>>(dstArr, cnt, E);

  if (tilesFull > 0)
    enc_edge_kernel<<<edgeBlocks, 256, 0, stream>>>(srcArr, dstArr, P, Q, encBt, encB2,
                                                    sumEnc, tilesFull);
  if (rem > 0)
    enc_tail_kernel<<<1, 256, 0, stream>>>(srcArr, dstArr, P, Q, encW2, encB2,
                                           sumEnc, eTail, E);

  node_update_kernel<<<(N + 255) / 256, 256, 0, stream>>>(sumEnc, cnt, eps, muW, muB,
                                                          varW, varB, decW1, decB1,
                                                          outMu, outLV, Pd, Qd, N);

  if (tilesFull > 0)
    dec_edge_kernel<<<edgeBlocks, 256, 0, stream>>>(srcArr, dstArr, Pd, Qd, decBt, decB2,
                                                    dec3Bt, decB3, sumDec, tilesFull);
  if (rem > 0)
    dec_tail_kernel<<<1, 256, 0, stream>>>(srcArr, dstArr, Pd, Qd, decW2, decB2,
                                           decW3, decB3, sumDec, eTail, E);

  finalize_kernel<<<(N * 4 + 255) / 256, 256, 0, stream>>>(sumDec, cnt, outO, N);
}
